// CrossAttention_42125039239237
// MI455X (gfx1250) — matrix operand order repaired and checked by reading, not compiled, not run on hardware
//
#include <hip/hip_runtime.h>

#define BATCH   4
#define TLEN    2048
#define SLEN    4096
#define NHEADS  8
#define DHEAD   32
#define DMODEL  256
// softmax scale folded into Q, in log2 domain: 32^-0.5 * log2(e)
#define QSCALE  0.25502763f

#define NSB     128          // attention S-block width
#define NSUB    (NSB / 16)   // 8 score subtiles
#define NKB     (NSB / 32)   // 4 K-dim blocks for P@V

typedef __attribute__((ext_vector_type(16))) __bf16 v16bf;
typedef __attribute__((ext_vector_type(8)))  float  v8f;
typedef __attribute__((ext_vector_type(4)))  unsigned int v4u;
typedef __attribute__((ext_vector_type(8)))  int    v8i;
typedef __attribute__((ext_vector_type(4)))  int    v4i;

// ---------------------------------------------------------------------------
// WMMA helper: D = A(16x32 bf16) * B(32x16 bf16) + C(16x16 f32)
// ---------------------------------------------------------------------------
__device__ __forceinline__ v8f wmma_bf16(v16bf a, v16bf b, v8f c) {
  return __builtin_amdgcn_wmma_f32_16x16x32_bf16(
      false, a, false, b, (short)0, c, false, false);
}

// A-operand loader from f32 row-major [M][K], converting to bf16.
// Contiguous 16-bit A 16x32 order used here: lane L holds M=L&15; element j -> K =
//   j + 16*(L>>4)   (pairs contiguous -> float2 loads)
__device__ __forceinline__ v16bf load_A_f32(const float* __restrict__ base,
                                            int ld, int m0, int k0, int lane) {
  const float* p = base + (size_t)(m0 + (lane & 15)) * ld + k0 + ((lane >> 4) << 4);
  v16bf a;
#pragma unroll
  for (int jp = 0; jp < 8; ++jp) {
    int kk = jp * 2;   // 0,2,4,...,14
    float2 f = *(const float2*)(p + kk);
    a[2 * jp]     = (__bf16)f.x;
    a[2 * jp + 1] = (__bf16)f.y;
  }
  return a;
}

// Same A layout from bf16 row-major [M][K].
__device__ __forceinline__ v16bf load_A_bf16(const __bf16* __restrict__ base,
                                             int ld, int m0, int k0, int lane) {
  const __bf16* p = base + (size_t)(m0 + (lane & 15)) * ld + k0 + ((lane >> 4) << 4);
  v16bf a;
#pragma unroll
  for (int jp = 0; jp < 8; ++jp) {
    int kk = jp * 2;
    a[2 * jp]     = p[kk];
    a[2 * jp + 1] = p[kk + 1];
  }
  return a;
}

// B-operand loader. BT = B transposed, row-major [N][K] bf16.
// Contiguous B 32x16 order used here (matches A): lanes 0-15: N=lane, K=0..15; lanes 16-31: K=16..31.
__device__ __forceinline__ v16bf load_BT(const __bf16* __restrict__ BT,
                                         int ld, int n0, int k0, int lane) {
  const __bf16* p = BT + (size_t)(n0 + (lane & 15)) * ld + k0 + ((lane >> 4) << 4);
  return *(const v16bf*)p;
}

// ---------------------------------------------------------------------------
// TDM: 2D tile (bf16) from global -> LDS via tensor_load_to_lds.
// D# packing per CDNA5 ISA (count=1, type=2,
// data_size=2B, no pad/iterate/gather, groups 2&3 zero => 2D tensor).
// 6-arg builtin form (clang-23 / therock-10.0 toolchain).
// ---------------------------------------------------------------------------
__device__ __forceinline__ void tdm_load_2d(unsigned int lds_off, const void* gptr,
                                            unsigned int tensor_d0, unsigned int tensor_d1,
                                            unsigned long long stride0,
                                            unsigned int tile_d0, unsigned int tile_d1) {
  unsigned long long ga = (unsigned long long)(uintptr_t)gptr;
  v4u g0;
  g0[0] = 1u;                                   // count=1 (valid user descriptor)
  g0[1] = lds_off;                              // LDS byte address
  g0[2] = (unsigned int)ga;                     // global_addr[31:0]
  g0[3] = (unsigned int)((ga >> 32) & 0x01FFFFFFu) | (2u << 30);  // addr[56:32] | type=2
  v8i g1;
  g1[0] = (int)(1u << 16);                      // workgroup_mask=0, data_size=1 (2B)
  g1[1] = (int)((tensor_d0 & 0xFFFFu) << 16);   // tensor_dim0[15:0]
  g1[2] = (int)((tensor_d0 >> 16) | ((tensor_d1 & 0xFFFFu) << 16));
  g1[3] = (int)((tensor_d1 >> 16) | (tile_d0 << 16));
  g1[4] = (int)tile_d1;                         // tile_dim2 = 0 (2D)
  g1[5] = (int)(unsigned int)(stride0 & 0xFFFFFFFFull);
  g1[6] = (int)(unsigned int)((stride0 >> 32) & 0xFFFFull);  // dim1_stride=0 (unused)
  g1[7] = 0;
  v4i z4 = {0, 0, 0, 0};
  v8i z8 = {0, 0, 0, 0, 0, 0, 0, 0};
  __builtin_amdgcn_tensor_load_to_lds(g0, g1, z4, z4, z8, 0);
}

__device__ __forceinline__ unsigned int lds_offset_of(const void* p) {
  return (unsigned int)(uintptr_t)p;   // low 32 bits of LDS aperture addr = offset
}

// ---------------------------------------------------------------------------
// Weight transpose + f32 -> bf16:  W[k][n] (f32) -> Wt[n][k] (bf16)
// ---------------------------------------------------------------------------
__global__ void xpose_bf16_kernel(const float* __restrict__ W, __bf16* __restrict__ Wt,
                                  int K, int N) {
  int idx = blockIdx.x * blockDim.x + threadIdx.x;
  if (idx < K * N) {
    int n = idx / K;
    int k = idx - n * K;
    Wt[idx] = (__bf16)W[(size_t)k * N + n];
  }
}

// ---------------------------------------------------------------------------
// Q projection (pre-scaled by QSCALE for exp2-domain softmax)
// ---------------------------------------------------------------------------
__global__ void gemm_q_kernel(const float* __restrict__ X,
                              const __bf16* __restrict__ WtQ,
                              __bf16* __restrict__ Qbf) {
  int lane = threadIdx.x & 31;
  int w    = threadIdx.x >> 5;
  int m0   = blockIdx.x * 64 + w * 16;
  int n0   = blockIdx.y * 64;
  v8f acc[4] = {};
  for (int k0 = 0; k0 < DMODEL; k0 += 32) {
    v16bf a = load_A_f32(X, DMODEL, m0, k0, lane);
#pragma unroll
    for (int nt = 0; nt < 4; ++nt) {
      v16bf b = load_BT(WtQ, DMODEL, n0 + nt * 16, k0, lane);
      acc[nt] = wmma_bf16(a, b, acc[nt]);
    }
  }
  int r0 = m0 + ((lane >> 4) << 3);
  int cb = n0 + (lane & 15);
#pragma unroll
  for (int nt = 0; nt < 4; ++nt)
#pragma unroll
    for (int r = 0; r < 8; ++r)
      Qbf[(size_t)(r0 + r) * DMODEL + cb + nt * 16] = (__bf16)(acc[nt][r] * QSCALE);
}

// ---------------------------------------------------------------------------
// KV projection: cols <256 -> Kbf[B*S][256]; cols >=256 -> Vt[b][h][d][s]
// ---------------------------------------------------------------------------
__global__ void gemm_kv_kernel(const float* __restrict__ C,
                               const __bf16* __restrict__ WtKV,
                               __bf16* __restrict__ Kbf,
                               __bf16* __restrict__ Vt) {
  int lane = threadIdx.x & 31;
  int w    = threadIdx.x >> 5;
  int m0   = blockIdx.x * 64 + w * 16;
  int n0   = blockIdx.y * 64;
  v8f acc[4] = {};
  for (int k0 = 0; k0 < DMODEL; k0 += 32) {
    v16bf a = load_A_f32(C, DMODEL, m0, k0, lane);
#pragma unroll
    for (int nt = 0; nt < 4; ++nt) {
      v16bf b = load_BT(WtKV, DMODEL, n0 + nt * 16, k0, lane);
      acc[nt] = wmma_bf16(a, b, acc[nt]);
    }
  }
  int r0 = m0 + ((lane >> 4) << 3);
#pragma unroll
  for (int nt = 0; nt < 4; ++nt) {
    int col = n0 + nt * 16 + (lane & 15);
#pragma unroll
    for (int r = 0; r < 8; ++r) {
      int row = r0 + r;
      float v = acc[nt][r];
      if (col < DMODEL) {
        Kbf[(size_t)row * DMODEL + col] = (__bf16)v;
      } else {
        int c = col - DMODEL;            // h*32+d
        int b = row >> 12;               // row / SLEN
        int s = row & (SLEN - 1);
        Vt[((size_t)(b * (NHEADS * DHEAD) + c)) * SLEN + s] = (__bf16)v;
      }
    }
  }
}

// ---------------------------------------------------------------------------
// Flash attention with TDM-staged K/V tiles (double-buffered in LDS).
// Q[B][T][256] (pre-scaled), K[B][S][256], Vt[B][H][32][S] -> O[B][T][256]
// WG = 4 waves = 64 T-rows of one head; K/V shared via LDS.
// ---------------------------------------------------------------------------
__global__ void attn_kernel(const __bf16* __restrict__ Q,
                            const __bf16* __restrict__ K,
                            const __bf16* __restrict__ Vt,
                            __bf16* __restrict__ O) {
  __shared__ __align__(32) __bf16 kbuf[2][NSB][32];       // [s][d]
  __shared__ __align__(32) __bf16 vbuf[2][32][NSB];       // [d][s]
  __shared__ __align__(32) __bf16 plds[4][16][NSB + 8];   // per-wave P tile

  int lane = threadIdx.x & 31;
  int w    = threadIdx.x >> 5;

  int bid = blockIdx.x;                  // B * H * (T/64)
  int tb  = bid & 31;                    // T/64 = 32
  int h   = (bid >> 5) & (NHEADS - 1);
  int b   = bid >> 8;
  int t0  = tb * 64 + w * 16;

  const __bf16* Qh = Q  + (size_t)b * TLEN * DMODEL + h * DHEAD;
  const __bf16* Kh = K  + (size_t)b * SLEN * DMODEL + h * DHEAD;
  const __bf16* Vh = Vt + (size_t)(b * NHEADS + h) * DHEAD * SLEN;

  v16bf qa = load_A_bf16(Qh, DMODEL, t0, 0, lane);   // 16x32 Q tile

  float m_i[8], l_i[8];
  v8f acc0 = {}, acc1 = {};              // d columns 0..15 / 16..31
#pragma unroll
  for (int r = 0; r < 8; ++r) { m_i[r] = -1e30f; l_i[r] = 0.0f; }

  int col     = lane & 15;
  int khalf16 = (lane >> 4) << 4;
  int kh8     = (lane >> 4) << 3;

  // Prologue: wave 0 kicks TDM for buffer 0
  if (w == 0) {
    tdm_load_2d(lds_offset_of(&kbuf[0][0][0]), Kh, 32, SLEN, DMODEL, 32, NSB);
    tdm_load_2d(lds_offset_of(&vbuf[0][0][0]), Vh, SLEN, 32, SLEN, NSB, 32);
  }

  const int NITER = SLEN / NSB;          // 32
  for (int it = 0; it < NITER; ++it) {
    int bf = it & 1;
    if (w == 0) {
      // issue next block's TDM (dummy reload of block 0 on the last iter),
      // then wait until only those 2 remain outstanding -> current buf ready
      int snext = (it + 1 < NITER) ? (it + 1) * NSB : 0;
      int nbf = bf ^ 1;
      tdm_load_2d(lds_offset_of(&kbuf[nbf][0][0]),
                  Kh + (size_t)snext * DMODEL, 32, SLEN, DMODEL, 32, NSB);
      tdm_load_2d(lds_offset_of(&vbuf[nbf][0][0]),
                  Vh + snext, SLEN, 32, SLEN, NSB, 32);
      __builtin_amdgcn_s_wait_tensorcnt(2);
    }
    __syncthreads();                     // buffer bf ready for all waves

    // ---- scores: 8 WMMAs of Q(16x32) x K^T(32x16) from LDS ----
    v8f sc[NSUB];
#pragma unroll
    for (int nt = 0; nt < NSUB; ++nt) {
      v16bf kb = *(const v16bf*)&kbuf[bf][nt * 16 + col][khalf16];
      v8f z = {};
      sc[nt] = wmma_bf16(qa, kb, z);
    }

    // ---- online softmax in exp2 domain (scale already folded into Q) ----
#pragma unroll
    for (int r = 0; r < 8; ++r) {
      float mx = sc[0][r];
#pragma unroll
      for (int nt = 1; nt < NSUB; ++nt) mx = fmaxf(mx, sc[nt][r]);
#pragma unroll
      for (int off = 1; off < 16; off <<= 1)
        mx = fmaxf(mx, __shfl_xor(mx, off, 32));
      float mnew  = fmaxf(m_i[r], mx);
      float alpha = __builtin_exp2f(m_i[r] - mnew);
      m_i[r] = mnew;
      float rowsum = 0.0f;
#pragma unroll
      for (int nt = 0; nt < NSUB; ++nt) {
        float p = __builtin_exp2f(sc[nt][r] - mnew);
        sc[nt][r] = p;
        rowsum += p;
      }
#pragma unroll
      for (int off = 1; off < 16; off <<= 1)
        rowsum += __shfl_xor(rowsum, off, 32);
      l_i[r] = l_i[r] * alpha + rowsum;
      acc0[r] *= alpha;
      acc1[r] *= alpha;
    }

    // ---- P: C-layout -> LDS [row][s_off] ----
    int rowbase = kh8;
#pragma unroll
    for (int nt = 0; nt < NSUB; ++nt)
#pragma unroll
      for (int r = 0; r < 8; ++r)
        plds[w][rowbase + r][nt * 16 + col] = (__bf16)sc[nt][r];
    __syncthreads();

    // ---- O += P(16x128) x V(128x32): 8 WMMAs ----
#pragma unroll
    for (int kb = 0; kb < NKB; ++kb) {
      v16bf pa;
#pragma unroll
      for (int jp = 0; jp < 8; ++jp) {
        int kk = kb * 32 + khalf16 + jp * 2;
        pa[2 * jp]     = plds[w][col][kk];
        pa[2 * jp + 1] = plds[w][col][kk + 1];
      }
      acc0 = wmma_bf16(pa, *(const v16bf*)&vbuf[bf][col][kb * 32 + khalf16], acc0);
      acc1 = wmma_bf16(pa, *(const v16bf*)&vbuf[bf][16 + col][kb * 32 + khalf16], acc1);
    }
    __syncthreads();   // all reads of buffers done before next TDM overwrites
  }

  // ---- epilogue: normalize and store bf16 O tile ----
  int rbase = t0 + kh8;
  __bf16* Oh = O + (size_t)b * TLEN * DMODEL + h * DHEAD;
#pragma unroll
  for (int r = 0; r < 8; ++r) {
    float inv = 1.0f / l_i[r];
    Oh[(size_t)(rbase + r) * DMODEL + col]      = (__bf16)(acc0[r] * inv);
    Oh[(size_t)(rbase + r) * DMODEL + 16 + col] = (__bf16)(acc1[r] * inv);
  }
}

// ---------------------------------------------------------------------------
// Output projection: out[B*T][256] = O16 @ w_out + b_out   (f32 out)
// ---------------------------------------------------------------------------
__global__ void gemm_out_kernel(const __bf16* __restrict__ O16,
                                const __bf16* __restrict__ WtO,
                                const float* __restrict__ bout,
                                float* __restrict__ out) {
  int lane = threadIdx.x & 31;
  int w    = threadIdx.x >> 5;
  int m0   = blockIdx.x * 64 + w * 16;
  int n0   = blockIdx.y * 64;
  v8f acc[4] = {};
  for (int k0 = 0; k0 < DMODEL; k0 += 32) {
    v16bf a = load_A_bf16(O16, DMODEL, m0, k0, lane);
#pragma unroll
    for (int nt = 0; nt < 4; ++nt) {
      v16bf b = load_BT(WtO, DMODEL, n0 + nt * 16, k0, lane);
      acc[nt] = wmma_bf16(a, b, acc[nt]);
    }
  }
  int r0 = m0 + ((lane >> 4) << 3);
#pragma unroll
  for (int nt = 0; nt < 4; ++nt) {
    int c = n0 + nt * 16 + (lane & 15);
    float bb = bout[c];
#pragma unroll
    for (int r = 0; r < 8; ++r)
      out[(size_t)(r0 + r) * DMODEL + c] = acc[nt][r] + bb;
  }
}

// ---------------------------------------------------------------------------
extern "C" void kernel_launch(void* const* d_in, const int* in_sizes, int n_in,
                              void* d_out, int out_size, void* d_ws, size_t ws_size,
                              hipStream_t stream) {
  const float* x     = (const float*)d_in[0];
  const float* ctx   = (const float*)d_in[1];
  const float* w_q   = (const float*)d_in[2];
  const float* w_kv  = (const float*)d_in[3];
  const float* w_out = (const float*)d_in[4];
  const float* b_out = (const float*)d_in[5];
  float* out = (float*)d_out;

  char* ws = (char*)d_ws;
  size_t off = 0;
  auto wsalloc = [&](size_t bytes) -> void* {
    void* p = ws + off;
    off += (bytes + 255) & ~(size_t)255;
    return p;
  };
  __bf16* Qbf  = (__bf16*)wsalloc((size_t)BATCH * TLEN * DMODEL * 2);
  __bf16* Kbf  = (__bf16*)wsalloc((size_t)BATCH * SLEN * DMODEL * 2);
  __bf16* Vt   = (__bf16*)wsalloc((size_t)BATCH * SLEN * DMODEL * 2);
  __bf16* O16  = (__bf16*)wsalloc((size_t)BATCH * TLEN * DMODEL * 2);
  __bf16* WtQ  = (__bf16*)wsalloc((size_t)DMODEL * DMODEL * 2);
  __bf16* WtKV = (__bf16*)wsalloc((size_t)DMODEL * 2 * DMODEL * 2);
  __bf16* WtO  = (__bf16*)wsalloc((size_t)DMODEL * DMODEL * 2);

  xpose_bf16_kernel<<<dim3(256), 256, 0, stream>>>(w_q, WtQ, DMODEL, DMODEL);
  xpose_bf16_kernel<<<dim3(512), 256, 0, stream>>>(w_kv, WtKV, DMODEL, 2 * DMODEL);
  xpose_bf16_kernel<<<dim3(256), 256, 0, stream>>>(w_out, WtO, DMODEL, DMODEL);

  gemm_q_kernel<<<dim3(128, 4), 128, 0, stream>>>(x, WtQ, Qbf);
  gemm_kv_kernel<<<dim3(256, 8), 128, 0, stream>>>(ctx, WtKV, Kbf, Vt);
  attn_kernel<<<dim3(BATCH * NHEADS * (TLEN / 64)), 128, 0, stream>>>(Qbf, Kbf, Vt, O16);
  gemm_out_kernel<<<dim3(128, 4), 128, 0, stream>>>(O16, WtO, b_out, out);
}
